// MoELayerOptimized_8211977470389
// MI455X (gfx1250) — matrix operand order repaired and checked by reading, not compiled, not run on hardware
//
#include <hip/hip_runtime.h>
#include <hip/hip_bf16.h>
#include <math.h>

// ---------------- types ----------------
typedef __bf16 bf16_t;
typedef __attribute__((ext_vector_type(16))) __bf16 v16bf;
typedef __attribute__((ext_vector_type(8)))  __bf16 bf16x8;
typedef __attribute__((ext_vector_type(8)))  float  v8f;

// ---------------- problem constants ----------------
static constexpr int D    = 1024;
static constexpr int H    = 4096;
static constexpr int E    = 8;
static constexpr int NTOK = 4096;          // B*T
static constexpr int TOPK = 2;
static constexpr int NA   = NTOK * TOPK;   // 8192 assignments
static constexpr int MT   = 128;           // M tile (slots)
static constexpr int NT   = 128;           // N tile (cols)
static constexpr int KT   = 32;            // bf16 WMMA K depth
static constexpr int CAP  = NA + E * MT;   // 9216 padded slot capacity
static constexpr int LDA  = KT + 8;        // LDS row stride (elements), 80B = 5x16B

// ---------------- workspace layout (bytes); total ~72.2 MB ----------------
static constexpr size_t WS_TOPK_I = 0;                       // int[NA]
static constexpr size_t WS_TOPK_W = 32768;                   // float[NA]
static constexpr size_t WS_CNT    = 65536;                   // int[E]
static constexpr size_t WS_OFF    = 65792;                   // int[E+1]
static constexpr size_t WS_CUR    = 66048;                   // int[E]
static constexpr size_t WS_ATOK   = 66304;                   // int[CAP]   (36864 B)
static constexpr size_t WS_AWT    = 103168;                  // float[CAP] (36864 B)
static constexpr size_t WS_H      = 140032;                  // bf16[CAP*H] (256-aligned)

// ---------------- helpers ----------------
__device__ __forceinline__ unsigned bfbits(float f) {
    union { float f; unsigned u; } a; a.f = f;
    return (a.u + 0x7FFFu + ((a.u >> 16) & 1u)) >> 16;       // RNE
}
__device__ __forceinline__ unsigned pack2bf(float lo, float hi) {
    return bfbits(lo) | (bfbits(hi) << 16);
}
__device__ __forceinline__ bf16_t f2bf(float f) {
    union { unsigned short s; bf16_t b; } o; o.s = (unsigned short)bfbits(f);
    return o.b;
}
__device__ __forceinline__ float gelu_exact(float v) {
    return 0.5f * v * (1.0f + erff(v * 0.70710678118654752f));
}

// ---------------- kernel 1: init (zero out, counts; pad tokens = -1) --------
__global__ void moe_init_kernel(float* __restrict__ out, int* __restrict__ counts,
                                int* __restrict__ atok) {
    int i = blockIdx.x * blockDim.x + threadIdx.x;
    if (i < NTOK * D) out[i] = 0.0f;
    if (i < CAP) atok[i] = -1;
    if (i < E) counts[i] = 0;
}

// ---------------- kernel 2: gating (logits, top-2, softmax, counts) ---------
__global__ void moe_gate_kernel(const float* __restrict__ x,
                                const float* __restrict__ gw,
                                const float* __restrict__ gb,
                                int* __restrict__ topk_i, float* __restrict__ topk_w,
                                int* __restrict__ counts) {
    __shared__ float s_gw[D * E];      // 32 KB
    __shared__ float s_log[32][E];
    int tid = threadIdx.x;
    for (int j = tid; j < D * E; j += 256) s_gw[j] = gw[j];
    __syncthreads();

    int tl = tid >> 3, e = tid & 7;
    int token = blockIdx.x * 32 + tl;
    const float4* xr = (const float4*)(x + (size_t)token * D);
    float acc = 0.0f;
    for (int q = 0; q < D / 4; ++q) {
        float4 xv = xr[q];
        const float* g = &s_gw[(q * 4) * E + e];
        acc += xv.x * g[0] + xv.y * g[E] + xv.z * g[2 * E] + xv.w * g[3 * E];
    }
    s_log[tl][e] = acc + gb[e];
    __syncthreads();

    if (tid < 32) {
        int n = blockIdx.x * 32 + tid;
        float l[E];
        for (int j = 0; j < E; ++j) l[j] = s_log[tid][j];
        int i0 = 0; float v0 = l[0];
        for (int j = 1; j < E; ++j) if (l[j] > v0) { v0 = l[j]; i0 = j; }
        int i1 = (i0 == 0) ? 1 : 0; float v1 = l[i1];
        for (int j = 0; j < E; ++j) { if (j == i0 || j == i1) continue; if (l[j] > v1) { v1 = l[j]; i1 = j; } }
        float eb = expf(v1 - v0);               // v1 <= v0 : stable
        float p0 = 1.0f / (1.0f + eb);
        float p1 = eb / (1.0f + eb);
        topk_i[n * 2 + 0] = i0; topk_i[n * 2 + 1] = i1;
        topk_w[n * 2 + 0] = p0; topk_w[n * 2 + 1] = p1;
        atomicAdd(&counts[i0], 1);
        atomicAdd(&counts[i1], 1);
    }
}

// ---------------- kernel 3: prefix scan of padded expert offsets ------------
__global__ void moe_scan_kernel(const int* __restrict__ counts,
                                int* __restrict__ offs, int* __restrict__ cur) {
    if (threadIdx.x == 0) {
        int off = 0;
        for (int e = 0; e < E; ++e) {
            offs[e] = off; cur[e] = off;
            off += ((counts[e] + MT - 1) / MT) * MT;   // pad to M tile
        }
        offs[E] = off;
    }
}

// ---------------- kernel 4: scatter assignments by expert -------------------
__global__ void moe_scatter_kernel(const int* __restrict__ topk_i,
                                   const float* __restrict__ topk_w,
                                   int* __restrict__ cur,
                                   int* __restrict__ atok, float* __restrict__ awt) {
    int n = blockIdx.x * blockDim.x + threadIdx.x;
    if (n >= NTOK) return;
    for (int k = 0; k < TOPK; ++k) {
        int e = topk_i[n * 2 + k];
        int slot = atomicAdd(&cur[e], 1);
        atok[slot] = n;
        awt[slot]  = topk_w[n * 2 + k];
    }
}

// ---------------- WMMA fragment loads from LDS ------------------------------
__device__ __forceinline__ v16bf load_frag_A(const bf16_t* sA, int row, int half) {
    // A 16x32 bf16: lanes 0-15: K {0..7,16..23}; lanes 16-31: K {8..15,24..31}
    int kb = half ? 8 : 0;
    bf16x8 lo = *(const bf16x8*)&sA[row * LDA + kb];
    bf16x8 hi = *(const bf16x8*)&sA[row * LDA + kb + 16];
    return __builtin_shufflevector(lo, hi, 0,1,2,3,4,5,6,7,8,9,10,11,12,13,14,15);
}
__device__ __forceinline__ v16bf load_frag_B(const bf16_t* sB, int col, int half) {
    // B 32x16 bf16 (K-contig in LDS rows): lanes 0-15: K {0..7,16..23}; lanes 16-31: K {8..15,24..31}
    int kb = half ? 8 : 0;
    bf16x8 lo = *(const bf16x8*)&sB[col * LDA + kb];
    bf16x8 hi = *(const bf16x8*)&sB[col * LDA + kb + 16];
    return __builtin_shufflevector(lo, hi, 0,1,2,3,4,5,6,7,8,9,10,11,12,13,14,15);
}

// Stage a KTxNT fp32 weight tile transposed into LDS as bf16 (K-contiguous rows).
// 16 consecutive lanes sweep contiguous N (coalesced reads); (k,k+1) pairs are
// packed into one u32 -> ds_store_b32.
__device__ __forceinline__ void stage_B_tile(bf16_t* sB, const float* __restrict__ wrow,
                                             size_t ldw, int tid) {
    int kp = (tid >> 4) * 2;          // even k: 0..30
    int nb = (tid & 15) * 8;          // n base: 0..120
    const float* w0 = wrow + (size_t)kp * ldw + nb;
    const float* w1 = w0 + ldw;
    float4 a0 = *(const float4*)w0;
    float4 a1 = *(const float4*)(w0 + 4);
    float4 b0 = *(const float4*)w1;
    float4 b1 = *(const float4*)(w1 + 4);
    float rl[8] = { a0.x, a0.y, a0.z, a0.w, a1.x, a1.y, a1.z, a1.w };
    float rh[8] = { b0.x, b0.y, b0.z, b0.w, b1.x, b1.y, b1.z, b1.w };
    unsigned* sB32 = (unsigned*)sB;
    int kh = kp >> 1;
    #pragma unroll
    for (int jj = 0; jj < 8; ++jj)
        sB32[(nb + jj) * (LDA / 2) + kh] = pack2bf(rl[jj], rh[jj]);
}

// ---------------- kernel 5: GEMM1  h = gelu(Xg @ W1[e] + b1[e]) -------------
// 128x128 tile, 8 waves in 4(m) x 2(n); each wave: 32x64 = 2x4 WMMA tiles.
__global__ void __launch_bounds__(256)
moe_gemm1_kernel(const float* __restrict__ x, const float* __restrict__ w1,
                 const float* __restrict__ b1, const int* __restrict__ offs,
                 const int* __restrict__ atok, bf16_t* __restrict__ hbuf) {
    __shared__ __align__(16) bf16_t sA[MT * LDA];   // 10240 B
    __shared__ __align__(16) bf16_t sB[NT * LDA];   // 10240 B
    __shared__ int sTok[MT];

    int s0 = blockIdx.y * MT;
    if (s0 >= offs[E]) return;
    int e = 0;
    while (e < E - 1 && s0 >= offs[e + 1]) ++e;
    int n0 = blockIdx.x * NT;
    int tid = threadIdx.x;
    if (tid < MT) sTok[tid] = atok[s0 + tid];

    int wid = tid >> 5, lane = tid & 31;
    int wm = wid >> 1, wn = wid & 1;           // 4x2 wave grid
    int half = lane >> 4, lr = lane & 15;

    v8f acc[2][4];
    for (int i = 0; i < 2; ++i)
        for (int j = 0; j < 4; ++j)
            for (int q = 0; q < 8; ++q) acc[i][j][q] = 0.0f;

    int arow = tid >> 1, acol = (tid & 1) * 16;      // A staging: 16 floats/thread
    const float* w1e = w1 + (size_t)e * D * H;

    for (int k0 = 0; k0 < D; k0 += KT) {
        __syncthreads();
        // ---- stage A (gathered token rows, fp32 -> packed bf16) ----
        int tok = sTok[arow];
        uint4 p0 = {0u,0u,0u,0u}, p1 = {0u,0u,0u,0u};
        if (tok >= 0) {
            const float* xr = x + (size_t)tok * D + k0 + acol;
            float4 f0 = *(const float4*)xr;
            float4 f1 = *(const float4*)(xr + 4);
            float4 f2 = *(const float4*)(xr + 8);
            float4 f3 = *(const float4*)(xr + 12);
            p0.x = pack2bf(f0.x, f0.y); p0.y = pack2bf(f0.z, f0.w);
            p0.z = pack2bf(f1.x, f1.y); p0.w = pack2bf(f1.z, f1.w);
            p1.x = pack2bf(f2.x, f2.y); p1.y = pack2bf(f2.z, f2.w);
            p1.z = pack2bf(f3.x, f3.y); p1.w = pack2bf(f3.z, f3.w);
        }
        *(uint4*)&sA[arow * LDA + acol]     = p0;
        *(uint4*)&sA[arow * LDA + acol + 8] = p1;
        // ---- stage B transposed: sB[n][k] = w1[e][k0+k][n0+n] ----
        const float* wrow = w1e + (size_t)k0 * H + n0;
        if (k0 + KT < D) __builtin_prefetch(wrow + (size_t)KT * H, 0, 1);
        stage_B_tile(sB, wrow, H, tid);
        __syncthreads();
        // ---- WMMA: 2x4 tiles per wave ----
        v16bf afr[2], bfr[4];
        for (int i = 0; i < 2; ++i) afr[i] = load_frag_A(sA, wm * 32 + i * 16 + lr, half);
        for (int j = 0; j < 4; ++j) bfr[j] = load_frag_B(sB, wn * 64 + j * 16 + lr, half);
        for (int i = 0; i < 2; ++i)
            for (int j = 0; j < 4; ++j)
                acc[i][j] = __builtin_amdgcn_wmma_f32_16x16x32_bf16(
                    false, afr[i], false, bfr[j], (short)0, acc[i][j], false, false);
    }

    // ---- epilogue: gelu(+bias), store bf16 h ----
    const float* b1e = b1 + (size_t)e * H;
    int mh = half * 8;
    for (int i = 0; i < 2; ++i)
        for (int j = 0; j < 4; ++j) {
            int col = n0 + wn * 64 + j * 16 + lr;
            float bias = b1e[col];
            for (int v = 0; v < 8; ++v) {
                int slot = s0 + wm * 32 + i * 16 + v + mh;
                float val = gelu_exact(acc[i][j][v] + bias);
                hbuf[(size_t)slot * H + col] = f2bf(val);
            }
        }
}

// ---------------- kernel 6: GEMM2  out += wt*(h @ W2[e] + b2[e]) ------------
__global__ void __launch_bounds__(256)
moe_gemm2_kernel(const bf16_t* __restrict__ hbuf, const float* __restrict__ w2,
                 const float* __restrict__ b2, const int* __restrict__ offs,
                 const int* __restrict__ atok, const float* __restrict__ awt,
                 float* __restrict__ out) {
    __shared__ __align__(16) bf16_t sA[MT * LDA];
    __shared__ __align__(16) bf16_t sB[NT * LDA];
    __shared__ int   sTok[MT];
    __shared__ float sWt[MT];

    int s0 = blockIdx.y * MT;
    if (s0 >= offs[E]) return;
    int e = 0;
    while (e < E - 1 && s0 >= offs[e + 1]) ++e;
    int n0 = blockIdx.x * NT;
    int tid = threadIdx.x;
    if (tid < MT) { sTok[tid] = atok[s0 + tid]; sWt[tid] = awt[s0 + tid]; }

    int wid = tid >> 5, lane = tid & 31;
    int wm = wid >> 1, wn = wid & 1;
    int half = lane >> 4, lr = lane & 15;

    v8f acc[2][4];
    for (int i = 0; i < 2; ++i)
        for (int j = 0; j < 4; ++j)
            for (int q = 0; q < 8; ++q) acc[i][j][q] = 0.0f;

    int arow = tid >> 1, acol = (tid & 1) * 16;      // A tile: 128x32 bf16 = 8 KB
    const float* w2e = w2 + (size_t)e * H * D;

    // Async DMA plumbing for the A tile (bf16 global -> LDS, no transform):
    // per-lane wave-relative LDS byte address (low 32 bits of generic ptr) and
    // per-lane 32-bit offset against a uniform SGPR base. IOFFSET is added to
    // BOTH the LDS and memory addresses (ISA 10.7), so one VGPR pair serves
    // both 16-byte transfers.
    unsigned ldsA = (unsigned)(size_t)&sA[arow * LDA + acol];
    unsigned voff = (unsigned)((arow * H + acol) * sizeof(bf16_t));

    for (int k0 = 0; k0 < H; k0 += KT) {
        __syncthreads();
        // ---- stage A via CDNA5 async global->LDS DMA (ASYNCcnt tracked) ----
        unsigned long long sbase =
            (unsigned long long)(size_t)(hbuf + (size_t)s0 * H + k0);
        asm volatile("global_load_async_to_lds_b128 %0, %1, %2"
                     :: "v"(ldsA), "v"(voff), "s"(sbase) : "memory");
        asm volatile("global_load_async_to_lds_b128 %0, %1, %2 offset:16"
                     :: "v"(ldsA), "v"(voff), "s"(sbase) : "memory");
        // ---- stage B transposed: sB[n][k] = w2[e][k0+k][n0+n] ----
        const float* wrow = w2e + (size_t)k0 * D + n0;
        if (k0 + KT < H) __builtin_prefetch(wrow + (size_t)KT * D, 0, 1);
        stage_B_tile(sB, wrow, D, tid);
        asm volatile("s_wait_asynccnt 0x0" ::: "memory");
        __syncthreads();
        // ---- WMMA: 2x4 tiles per wave ----
        v16bf afr[2], bfr[4];
        for (int i = 0; i < 2; ++i) afr[i] = load_frag_A(sA, wm * 32 + i * 16 + lr, half);
        for (int j = 0; j < 4; ++j) bfr[j] = load_frag_B(sB, wn * 64 + j * 16 + lr, half);
        for (int i = 0; i < 2; ++i)
            for (int j = 0; j < 4; ++j)
                acc[i][j] = __builtin_amdgcn_wmma_f32_16x16x32_bf16(
                    false, afr[i], false, bfr[j], (short)0, acc[i][j], false, false);
    }

    // ---- epilogue: weighted atomic combine into out ----
    const float* b2e = b2 + (size_t)e * D;
    int mh = half * 8;
    for (int i = 0; i < 2; ++i)
        for (int j = 0; j < 4; ++j) {
            int col = n0 + wn * 64 + j * 16 + lr;
            float bias = b2e[col];
            for (int v = 0; v < 8; ++v) {
                int m = wm * 32 + i * 16 + v + mh;
                int tok = sTok[m];
                if (tok >= 0) {
                    float val = sWt[m] * (acc[i][j][v] + bias);
                    atomicAdd(&out[(size_t)tok * D + col], val);
                }
            }
        }
}

// ---------------- host launcher ----------------
extern "C" void kernel_launch(void* const* d_in, const int* in_sizes, int n_in,
                              void* d_out, int out_size, void* d_ws, size_t ws_size,
                              hipStream_t stream) {
    const float* x  = (const float*)d_in[0];
    const float* gw = (const float*)d_in[1];
    const float* gb = (const float*)d_in[2];
    const float* w1 = (const float*)d_in[3];
    const float* b1 = (const float*)d_in[4];
    const float* w2 = (const float*)d_in[5];
    const float* b2 = (const float*)d_in[6];
    float* out = (float*)d_out;

    char* ws = (char*)d_ws;
    int*    topk_i = (int*)   (ws + WS_TOPK_I);
    float*  topk_w = (float*) (ws + WS_TOPK_W);
    int*    counts = (int*)   (ws + WS_CNT);
    int*    offs   = (int*)   (ws + WS_OFF);
    int*    cur    = (int*)   (ws + WS_CUR);
    int*    atok   = (int*)   (ws + WS_ATOK);
    float*  awt    = (float*) (ws + WS_AWT);
    bf16_t* hbuf   = (bf16_t*)(ws + WS_H);

    moe_init_kernel<<<(NTOK * D + 255) / 256, 256, 0, stream>>>(out, counts, atok);
    moe_gate_kernel<<<NTOK / 32, 256, 0, stream>>>(x, gw, gb, topk_i, topk_w, counts);
    moe_scan_kernel<<<1, 32, 0, stream>>>(counts, offs, cur);
    moe_scatter_kernel<<<NTOK / 256, 256, 0, stream>>>(topk_i, topk_w, cur, atok, awt);
    moe_gemm1_kernel<<<dim3(H / NT, CAP / MT), 256, 0, stream>>>(x, w1, b1, offs, atok, hbuf);
    moe_gemm2_kernel<<<dim3(D / NT, CAP / MT), 256, 0, stream>>>(hbuf, w2, b2, offs, atok, awt, out);
}
